// SheafNeuralNetwork_81698867905246
// MI455X (gfx1250) — compile-verified, matrix operand order checked by reading, not run on hardware
//
#include <hip/hip_runtime.h>

typedef __attribute__((ext_vector_type(16))) _Float16 v16h;
typedef __attribute__((ext_vector_type(8)))  _Float16 v8h;
typedef __attribute__((ext_vector_type(8)))  float    v8f;

#define E_EDGES 4096
#define FEAT    128
#define H0      32
#define H1      64
#define DSTALK  2
#define NROWS   (E_EDGES * DSTALK)   // 8192
#define MLPH    128
#define NCLS    2

// ---- WMMA helpers -----------------------------------------------------------
__device__ __forceinline__ v8f wmma_f16(v16h a, v16h b, v8f c) {
  // D = A(16x32 f16) x B(32x16 f16) + C(16x16 f32)
  return __builtin_amdgcn_wmma_f32_16x16x32_f16(
      /*neg_a=*/false, a, /*neg_b=*/false, b,
      /*c_mod=*/(short)0, c, /*reuse_a=*/false, /*reuse_b=*/false);
}

// f16 A/B fragment element->K mapping for 16x16x32 (wave32):
// element i (0..15), lane half hh: K = (i>>3)*16 + hh*8 + (i&7)
__device__ __forceinline__ int frag_k(int i, int hh) {
  return ((i >> 3) << 4) + (hh << 3) + (i & 7);
}

// A fragment = two contiguous 8-half chunks, 16 halves apart.
// p must point to (row base + hh*8 + ktile*32); 16B aligned.
__device__ __forceinline__ v16h frag_from(const _Float16* p) {
  v8h lo = *(const v8h*)(p);
  v8h hi = *(const v8h*)(p + 16);
  v16h r;
#pragma unroll
  for (int i = 0; i < 8; ++i) { r[i] = lo[i]; r[i + 8] = hi[i]; }
  return r;
}

// B fragment from pre-swizzled weights: 16 contiguous halves per lane.
__device__ __forceinline__ v16h bfrag(const _Float16* wsw, int blk, int lane) {
  return *(const v16h*)(wsw + (((size_t)blk * 32 + lane) << 4));
}

// ---- utility kernels --------------------------------------------------------
__global__ void cvt_f32_to_f16_kernel(const float* __restrict__ src,
                                      _Float16* __restrict__ dst, int n) {
  int i = blockIdx.x * blockDim.x + threadIdx.x;
  if (i < n) dst[i] = (_Float16)src[i];
}

__global__ void zero_f32_kernel(float* __restrict__ p, int n) {
  int i = blockIdx.x * blockDim.x + threadIdx.x;
  if (i < n) p[i] = 0.0f;
}

// Swizzle a (Kdim x Ndim) f32 weight into WMMA B-fragment order (f16):
// out[((kt*ntiles + nt)*32 + lane)*16 + i] = W[(kt*32 + frag_k(i,hh))*Ndim + nt*16 + n]
__global__ void swizzle_w_kernel(const float* __restrict__ W,
                                 _Float16* __restrict__ out, int Kdim, int Ndim) {
  int t = blockIdx.x * blockDim.x + threadIdx.x;
  int total = (Kdim >> 5) * (Ndim >> 4) * 32;
  if (t >= total) return;
  int lane = t & 31;
  int blk = t >> 5;
  int ntiles = Ndim >> 4;
  int kt = blk / ntiles, nt = blk % ntiles;
  int n = lane & 15, hh = lane >> 4;
  _Float16* dst = out + ((size_t)t << 4);
#pragma unroll
  for (int i = 0; i < 16; ++i) {
    int K = kt * 32 + frag_k(i, hh);
    dst[i] = (_Float16)W[(size_t)K * Ndim + nt * 16 + n];
  }
}

// ---- Xemb16 = f16(edge_features @ emb_W + emb_b)  (4096x128 @ 128x32) -------
__global__ __launch_bounds__(256) void emb_kernel(
    const _Float16* __restrict__ ef16,     // E x 128
    const _Float16* __restrict__ embWsw,   // swizzled 128x32
    const float* __restrict__ emb_b,       // 32
    _Float16* __restrict__ Xemb16) {       // E x 32
  const int wave = threadIdx.x >> 5;
  const int lane = threadIdx.x & 31;
  const int row0 = (blockIdx.x * 8 + wave) * 16;  // 256 tiles total
  const int m = lane & 15, hh = lane >> 4;

  const _Float16* arow = ef16 + (size_t)(row0 + m) * FEAT + hh * 8;
  v8f acc[2] = {v8f{}, v8f{}};
#pragma unroll
  for (int kt = 0; kt < 4; ++kt) {      // K = 128
    v16h a = frag_from(arow + kt * 32);
#pragma unroll
    for (int nt = 0; nt < 2; ++nt)      // N = 32
      acc[nt] = wmma_f16(a, bfrag(embWsw, kt * 2 + nt, lane), acc[nt]);
  }
#pragma unroll
  for (int nt = 0; nt < 2; ++nt)
#pragma unroll
    for (int r = 0; r < 8; ++r) {
      int row = row0 + hh * 8 + r;
      int col = nt * 16 + m;
      Xemb16[(size_t)row * H0 + col] = (_Float16)(acc[nt][r] + emb_b[col]);
    }
}

// ---- fused pair MLP: pf -> LN/ReLU -> LN/ReLU -> fmaps; deg += fmaps^2 ------
__global__ __launch_bounds__(128) void pair_mlp_kernel(
    const _Float16* __restrict__ ef16,   // E x 128
    const _Float16* __restrict__ m1sw,   // swizzled 256x64
    const float* __restrict__ m1b, const float* __restrict__ ln1g,
    const float* __restrict__ ln1b,
    const _Float16* __restrict__ m2sw,   // swizzled 64x32
    const float* __restrict__ m2b, const float* __restrict__ ln2g,
    const float* __restrict__ ln2b,
    const float* __restrict__ m3W,       // 32 x 2
    const float* __restrict__ m3b,       // 2
    const int* __restrict__ pair_i, const int* __restrict__ pair_j,
    float* __restrict__ fmaps,           // P x 2
    float* __restrict__ deg,             // E x 2 (pre-zeroed)
    int P) {
  __shared__ __align__(32) _Float16 sA[4][16][256];  // 32 KB: pf, then h1(f16)
  __shared__ __align__(16) float    sH[4][16][64];   // 16 KB: rows for LayerNorm

  const int wave = threadIdx.x >> 5;
  const int lane = threadIdx.x & 31;
  const int row0 = (blockIdx.x * 4 + wave) * 16;
  const int m = lane & 15, hh = lane >> 4;

  // ---- stage pf = [ef[i], ef[j]] as f16 rows (16 x 256), 16B per lane per row
  {
    const int half = hh;                 // 0: ef[pair_i] part, 1: ef[pair_j] part
    const int cc = (lane & 15) * 8;      // 8 halves
    for (int r = 0; r < 16; ++r) {
      int p = row0 + r;
      bool ok = p < P;
      v8h v = {};
      if (ok) {
        int e = half ? pair_j[p] : pair_i[p];
        v = *(const v8h*)(ef16 + (size_t)e * FEAT + cc);
      }
      *(v8h*)&sA[wave][r][half * FEAT + cc] = v;
    }
  }
  __syncthreads();

  // ---- layer 1: (16x256) @ (256x64)
  const _Float16* arow1 = &sA[wave][m][hh * 8];
  v8f acc[4] = {v8f{}, v8f{}, v8f{}, v8f{}};
#pragma unroll
  for (int kt = 0; kt < 8; ++kt) {
    v16h a = frag_from(arow1 + kt * 32);
#pragma unroll
    for (int nt = 0; nt < 4; ++nt)
      acc[nt] = wmma_f16(a, bfrag(m1sw, kt * 4 + nt, lane), acc[nt]);
  }
#pragma unroll
  for (int nt = 0; nt < 4; ++nt)
#pragma unroll
    for (int r = 0; r < 8; ++r) {
      int row = hh * 8 + r, col = nt * 16 + m;
      sH[wave][row][col] = acc[nt][r] + m1b[col];
    }
  __syncthreads();

  // ---- LN1 + ReLU -> h1 (f16) into sA[..][0..63]
  if (lane < 16) {
    const float4* hr4 = (const float4*)sH[wave][lane];
    float mu = 0.f;
#pragma unroll
    for (int c = 0; c < 16; ++c) {
      float4 t = hr4[c];
      mu += t.x + t.y + t.z + t.w;
    }
    mu *= (1.0f / 64.0f);
    float var = 0.f;
#pragma unroll
    for (int c = 0; c < 16; ++c) {
      float4 t = hr4[c];
      float a0 = t.x - mu, a1 = t.y - mu, a2 = t.z - mu, a3 = t.w - mu;
      var += a0 * a0 + a1 * a1 + a2 * a2 + a3 * a3;
    }
    var *= (1.0f / 64.0f);
    float inv = rsqrtf(var + 1e-5f);
    float* hr = sH[wave][lane];
    for (int c = 0; c < 64; ++c) {
      float v = (hr[c] - mu) * inv * ln1g[c] + ln1b[c];
      sA[wave][lane][c] = (_Float16)(v > 0.f ? v : 0.f);
    }
  }
  __syncthreads();

  // ---- layer 2: (16x64) @ (64x32)
  const _Float16* arow2 = &sA[wave][m][hh * 8];
  v8f acc2[2] = {v8f{}, v8f{}};
#pragma unroll
  for (int kt = 0; kt < 2; ++kt) {
    v16h a = frag_from(arow2 + kt * 32);
#pragma unroll
    for (int nt = 0; nt < 2; ++nt)
      acc2[nt] = wmma_f16(a, bfrag(m2sw, kt * 2 + nt, lane), acc2[nt]);
  }
#pragma unroll
  for (int nt = 0; nt < 2; ++nt)
#pragma unroll
    for (int r = 0; r < 8; ++r) {
      int row = hh * 8 + r, col = nt * 16 + m;
      sH[wave][row][col] = acc2[nt][r] + m2b[col];
    }
  __syncthreads();

  // ---- LN2 + ReLU (in place, f32), then layer 3 (32 -> 2) + deg scatter
  if (lane < 16) {
    const float4* hr4 = (const float4*)sH[wave][lane];
    float mu = 0.f;
#pragma unroll
    for (int c = 0; c < 8; ++c) {
      float4 t = hr4[c];
      mu += t.x + t.y + t.z + t.w;
    }
    mu *= (1.0f / 32.0f);
    float var = 0.f;
#pragma unroll
    for (int c = 0; c < 8; ++c) {
      float4 t = hr4[c];
      float a0 = t.x - mu, a1 = t.y - mu, a2 = t.z - mu, a3 = t.w - mu;
      var += a0 * a0 + a1 * a1 + a2 * a2 + a3 * a3;
    }
    var *= (1.0f / 32.0f);
    float inv = rsqrtf(var + 1e-5f);
    float* hr = sH[wave][lane];
    float f0 = m3b[0], f1 = m3b[1];
    for (int c = 0; c < 32; ++c) {
      float v = (hr[c] - mu) * inv * ln2g[c] + ln2b[c];
      v = v > 0.f ? v : 0.f;
      f0 += v * m3W[c * 2 + 0];
      f1 += v * m3W[c * 2 + 1];
    }
    int p = row0 + lane;
    if (p < P) {
      fmaps[(size_t)p * 2 + 0] = f0;
      fmaps[(size_t)p * 2 + 1] = f1;
      int vi = pair_i[p];
      atomicAdd(&deg[vi * 2 + 0], f0 * f0);
      atomicAdd(&deg[vi * 2 + 1], f1 * f1);
    }
  }
}

// ---- s = 1/sqrt(max(deg+1e-3, 1e-4)) ---------------------------------------
__global__ void scale_kernel(const float* __restrict__ deg,
                             float* __restrict__ s, int n) {
  int i = blockIdx.x * blockDim.x + threadIdx.x;
  if (i < n) s[i] = rsqrtf(fmaxf(deg[i] + 1e-3f, 1e-4f));
}

// ---- Zt = Xemb @ W2 ; R = Xemb @ res_W + res_b  (4096x32 @ 32x64, twice) ----
__global__ __launch_bounds__(256) void z_kernel(
    const _Float16* __restrict__ Xemb16,  // E x 32
    const _Float16* __restrict__ W2sw,    // swizzled 32x64
    const _Float16* __restrict__ resWsw,  // swizzled 32x64
    const float* __restrict__ res_b,      // 64
    float* __restrict__ Zt,               // E x 64
    float* __restrict__ R) {              // E x 64
  const int wave = threadIdx.x >> 5;
  const int lane = threadIdx.x & 31;
  const int row0 = (blockIdx.x * 8 + wave) * 16;
  const int m = lane & 15, hh = lane >> 4;

  v16h a = frag_from(Xemb16 + (size_t)(row0 + m) * H0 + hh * 8);  // K = 32

  v8f az[4] = {v8f{}, v8f{}, v8f{}, v8f{}};
  v8f ar[4] = {v8f{}, v8f{}, v8f{}, v8f{}};
#pragma unroll
  for (int nt = 0; nt < 4; ++nt) {
    az[nt] = wmma_f16(a, bfrag(W2sw, nt, lane), az[nt]);
    ar[nt] = wmma_f16(a, bfrag(resWsw, nt, lane), ar[nt]);
  }
#pragma unroll
  for (int nt = 0; nt < 4; ++nt)
#pragma unroll
    for (int r = 0; r < 8; ++r) {
      int row = row0 + hh * 8 + r;
      int col = nt * 16 + m;
      Zt[(size_t)row * H1 + col] = az[nt][r];
      R[(size_t)row * H1 + col]  = ar[nt][r] + res_b[col];
    }
}

// ---- diffused := diag-term:  (s^2*deg)[v,k] * wsum[k] * Zt[e,:] ------------
__global__ void diag_kernel(const float* __restrict__ s,
                            const float* __restrict__ deg,
                            const float* __restrict__ Zt,
                            const float* __restrict__ W1,
                            float* __restrict__ diffused) {
  int i = blockIdx.x * blockDim.x + threadIdx.x;  // over NROWS*64
  if (i >= NROWS * H1) return;
  int c = i & 63;
  int n = i >> 6;
  int k = n & 1;
  int e = n >> 1;
  float sv = s[e * 2 + k];
  float dv = sv * sv * deg[e * 2 + k];
  float wsum = W1[k * 2 + 0] + W1[k * 2 + 1];  // X rows repeat across stalk dim
  diffused[i] = dv * wsum * Zt[(size_t)e * H1 + c];
}

// ---- off-diagonal scatter: diffused[i*2+k,:] += val * Xt[j*2+k,:] ----------
__global__ __launch_bounds__(256) void scatter_kernel(
    const float* __restrict__ fmaps, const float* __restrict__ s,
    const float* __restrict__ Zt, const float* __restrict__ W1,
    const int* __restrict__ pair_i, const int* __restrict__ pair_j,
    const int* __restrict__ rev_idx, float* __restrict__ diffused, int P) {
  int t = threadIdx.x;
  int job = blockIdx.x * 4 + (t >> 6);  // one (p,k) per 64 threads
  if (job >= P * 2) return;
  int c = t & 63;
  int p = job >> 1;
  int k = job & 1;
  int vi = pair_i[p], vj = pair_j[p];
  float val = -(fmaps[(size_t)rev_idx[p] * 2 + k] * fmaps[(size_t)p * 2 + k]) *
              s[vi * 2 + k] * s[vj * 2 + k];
  float wsum = W1[k * 2 + 0] + W1[k * 2 + 1];
  float xs = wsum * Zt[(size_t)vj * H1 + c];
  atomicAdd(&diffused[(size_t)(vi * 2 + k) * H1 + c], val * xs);
}

// ---- final rows + pooled column sums ---------------------------------------
__global__ __launch_bounds__(256) void final_pool_kernel(
    const float* __restrict__ R, const float* __restrict__ diffused,
    const float* __restrict__ eps_p, float* __restrict__ gsum) {
  __shared__ float sdata[256];
  int t = threadIdx.x;
  int c = t & 63;
  int n = blockIdx.x * 4 + (t >> 6);  // 2048 blocks x 4 rows = 8192 rows
  int k = n & 1;
  int e = n >> 1;
  float dv = diffused[(size_t)n * H1 + c];
  float el = dv > 0.f ? dv : (__expf(dv) - 1.0f);
  float v = (1.0f + eps_p[k]) * R[(size_t)e * H1 + c] - el;
  sdata[t] = v;
  __syncthreads();
  if (t < 64) {
    float tot = sdata[t] + sdata[t + 64] + sdata[t + 128] + sdata[t + 192];
    atomicAdd(&gsum[t], tot);
  }
}

// ---- classifier (single block) ---------------------------------------------
__global__ __launch_bounds__(128) void cls_kernel(
    const float* __restrict__ gsum, const float* __restrict__ c1W,
    const float* __restrict__ c1b, const float* __restrict__ c2W,
    const float* __restrict__ c2b, const float* __restrict__ c3W,
    const float* __restrict__ c3b, float* __restrict__ out) {
  __shared__ float g[H1], h1[MLPH], h2[MLPH / 2];
  int t = threadIdx.x;
  if (t < H1) g[t] = gsum[t] * (1.0f / (float)NROWS);
  __syncthreads();
  {
    float acc = c1b[t];
    for (int i = 0; i < H1; ++i) acc += g[i] * c1W[i * MLPH + t];
    h1[t] = acc > 0.f ? acc : 0.f;
  }
  __syncthreads();
  if (t < MLPH / 2) {
    float acc = c2b[t];
    for (int i = 0; i < MLPH; ++i) acc += h1[i] * c2W[i * (MLPH / 2) + t];
    h2[t] = acc > 0.f ? acc : 0.f;
  }
  __syncthreads();
  if (t < NCLS) {
    float acc = c3b[t];
    for (int i = 0; i < MLPH / 2; ++i) acc += h2[i] * c3W[i * NCLS + t];
    out[t] = acc;
  }
}

// ---- host-side orchestration ------------------------------------------------
extern "C" void kernel_launch(void* const* d_in, const int* in_sizes, int n_in,
                              void* d_out, int out_size, void* d_ws,
                              size_t ws_size, hipStream_t stream) {
  const float* ef     = (const float*)d_in[0];
  const float* emb_W  = (const float*)d_in[1];
  const float* emb_b  = (const float*)d_in[2];
  const float* m1_W   = (const float*)d_in[3];
  const float* m1_b   = (const float*)d_in[4];
  const float* ln1_g  = (const float*)d_in[5];
  const float* ln1_b  = (const float*)d_in[6];
  const float* m2_W   = (const float*)d_in[7];
  const float* m2_b   = (const float*)d_in[8];
  const float* ln2_g  = (const float*)d_in[9];
  const float* ln2_b  = (const float*)d_in[10];
  const float* m3_W   = (const float*)d_in[11];
  const float* m3_b   = (const float*)d_in[12];
  const float* W1     = (const float*)d_in[13];
  const float* W2     = (const float*)d_in[14];
  const float* res_W  = (const float*)d_in[15];
  const float* res_b  = (const float*)d_in[16];
  const float* eps_p  = (const float*)d_in[17];
  const float* c1_W   = (const float*)d_in[18];
  const float* c1_b   = (const float*)d_in[19];
  const float* c2_W   = (const float*)d_in[20];
  const float* c2_b   = (const float*)d_in[21];
  const float* c3_W   = (const float*)d_in[22];
  const float* c3_b   = (const float*)d_in[23];
  const int* pair_i   = (const int*)d_in[24];
  const int* pair_j   = (const int*)d_in[25];
  const int* rev_idx  = (const int*)d_in[26];
  const int P = in_sizes[24];

  size_t off = 0;
  auto carve = [&](size_t bytes) -> void* {
    void* p = (char*)d_ws + off;
    off += (bytes + 255) & ~(size_t)255;
    return p;
  };
  _Float16* ef16   = (_Float16*)carve((size_t)E_EDGES * FEAT * 2);
  _Float16* m1sw   = (_Float16*)carve(256 * 64 * 2);
  _Float16* m2sw   = (_Float16*)carve(64 * 32 * 2);
  _Float16* embWsw = (_Float16*)carve(128 * 32 * 2);
  _Float16* W2sw   = (_Float16*)carve(32 * 64 * 2);
  _Float16* resWsw = (_Float16*)carve(32 * 64 * 2);
  _Float16* Xemb16 = (_Float16*)carve((size_t)E_EDGES * H0 * 2);
  float* fmaps     = (float*)carve((size_t)P * 2 * 4);
  float* deg       = (float*)carve((size_t)E_EDGES * 2 * 4);
  float* sbuf      = (float*)carve((size_t)E_EDGES * 2 * 4);
  float* Zt        = (float*)carve((size_t)E_EDGES * H1 * 4);
  float* Rbuf      = (float*)carve((size_t)E_EDGES * H1 * 4);
  float* diffused  = (float*)carve((size_t)NROWS * H1 * 4);
  float* gsum      = (float*)carve(64 * 4);

  auto gs = [](int n) { return (n + 255) / 256; };

  // init: zero accumulators, f16 conversion + weight swizzles
  zero_f32_kernel<<<gs(E_EDGES * 2), 256, 0, stream>>>(deg, E_EDGES * 2);
  zero_f32_kernel<<<1, 64, 0, stream>>>(gsum, 64);
  cvt_f32_to_f16_kernel<<<gs(E_EDGES * FEAT), 256, 0, stream>>>(ef, ef16, E_EDGES * FEAT);
  swizzle_w_kernel<<<gs(1024), 256, 0, stream>>>(m1_W, m1sw, 256, 64);
  swizzle_w_kernel<<<gs(128), 256, 0, stream>>>(m2_W, m2sw, 64, 32);
  swizzle_w_kernel<<<gs(256), 256, 0, stream>>>(emb_W, embWsw, 128, 32);
  swizzle_w_kernel<<<gs(128), 256, 0, stream>>>(W2, W2sw, 32, 64);
  swizzle_w_kernel<<<gs(128), 256, 0, stream>>>(res_W, resWsw, 32, 64);

  // embedding (WMMA): 4096/16 = 256 row tiles, 8 waves/block -> 32 blocks
  emb_kernel<<<32, 256, 0, stream>>>(ef16, embWsw, emb_b, Xemb16);

  // fused pair MLP (WMMA): 64 pairs per 128-thread block
  pair_mlp_kernel<<<(P + 63) / 64, 128, 0, stream>>>(
      ef16, m1sw, m1_b, ln1_g, ln1_b, m2sw, m2_b, ln2_g, ln2_b, m3_W, m3_b,
      pair_i, pair_j, fmaps, deg, P);

  // s = rsqrt(max(deg+1e-3,1e-4))
  scale_kernel<<<gs(E_EDGES * 2), 256, 0, stream>>>(deg, sbuf, E_EDGES * 2);

  // Zt = Xemb@W2, R = Xemb@res_W + res_b (WMMA)
  z_kernel<<<32, 256, 0, stream>>>(Xemb16, W2sw, resWsw, res_b, Zt, Rbuf);

  // sparse Delta @ Xt: diagonal term then off-diagonal scatter-adds
  diag_kernel<<<gs(NROWS * H1), 256, 0, stream>>>(sbuf, deg, Zt, W1, diffused);
  scatter_kernel<<<(P * 2 + 3) / 4, 256, 0, stream>>>(
      fmaps, sbuf, Zt, W1, pair_i, pair_j, rev_idx, diffused, P);

  // final rows + mean pool
  final_pool_kernel<<<NROWS / 4, 256, 0, stream>>>(Rbuf, diffused, eps_p, gsum);

  // classifier head
  cls_kernel<<<1, 128, 0, stream>>>(gsum, c1_W, c1_b, c2_W, c2_b, c3_W, c3_b,
                                    (float*)d_out);
}
